// TripletMinerLoss_29858612642177
// MI455X (gfx1250) — compile-verified, matrix operand order checked by reading, not run on hardware
//
#include <hip/hip_runtime.h>
#include <stdint.h>

// ---------------------------------------------------------------------------
// TripletMinerLoss on MI455X (gfx1250):
//   dists = 1 - z_s @ z_i^T   (4096x4096, K=512)  via V_WMMA_F32_16X16X32_BF16
//   compensated bf16 split (hi+lo) for ~fp32 accuracy: 3 WMMAs per K-chunk.
//   4 column tiles in flight per wave: amortizes A-fragment LDS loads 4x and
//   gives 4 independent WMMA accumulation chains for latency hiding.
//   Mining (diag positive, masked row-min negative) fused into the GEMM
//   epilogue so the 64MB distance matrix is never written to memory.
// ---------------------------------------------------------------------------

typedef __attribute__((ext_vector_type(16))) __bf16 v16bf;
typedef __attribute__((ext_vector_type(8)))  __bf16 v8bf;
typedef __attribute__((ext_vector_type(8)))  float  v8f;

#define B_N        4096
#define D_K        512
#define NTILE      (B_N / 16)     // 256 column tiles of 16
#define NSTRIP     (B_N / 16)     // 256 row strips of 16
#define KCHUNKS    (D_K / 32)     // 16 bf16-WMMA K-chunks
#define TBLK       4              // column tiles in flight per wave
#define MARGIN_F   0.2f
#define BIG_F      1e9f
#define LDS_STRIDE 520            // 512 + 8 halves pad -> rows land on different banks

__device__ __forceinline__ v16bf cat16(v8bf lo, v8bf hi) {
    return __builtin_shufflevector(lo, hi, 0, 1, 2, 3, 4, 5, 6, 7,
                                           8, 9, 10, 11, 12, 13, 14, 15);
}

// --------------------------- kernel 0: zero accumulators --------------------
__global__ void tm_zero_accum(float* total, int* count) {
    *total = 0.0f;
    *count = 0;
}

// ------------------- kernel 1: fp32 -> (hi,lo) bf16 split -------------------
__global__ void tm_split_bf16(const float* __restrict__ zs,
                              const float* __restrict__ zi,
                              __bf16* __restrict__ zsh, __bf16* __restrict__ zsl,
                              __bf16* __restrict__ zih, __bf16* __restrict__ zil,
                              int n) {
    int i = blockIdx.x * blockDim.x + threadIdx.x;
    if (i < n) {
        float x = zs[i];
        __bf16 h = (__bf16)x;
        zsh[i] = h;
        zsl[i] = (__bf16)(x - (float)h);
    } else if (i < 2 * n) {
        int j = i - n;
        float x = zi[j];
        __bf16 h = (__bf16)x;
        zih[j] = h;
        zil[j] = (__bf16)(x - (float)h);
    }
}

// --------------- kernel 2: fused WMMA GEMM + triplet mining -----------------
// grid = 256 blocks (one 16-row strip each), block = 256 threads = 8 waves.
// Each wave processes TBLK=4 adjacent column tiles at a time.
__global__ __launch_bounds__(256, 1) void tm_gemm_mine(
    const float*  __restrict__ zs,      // fp32 z_s for LDS staging/split
    const __bf16* __restrict__ zih,
    const __bf16* __restrict__ zil,
    const int*    __restrict__ labels,
    float* __restrict__ total, int* __restrict__ count) {

    __shared__ __align__(16) __bf16 Ah[16 * LDS_STRIDE];
    __shared__ __align__(16) __bf16 Al[16 * LDS_STRIDE];
    __shared__ int   rowlab[16];
    __shared__ float pdiag[16];
    __shared__ float wmin[8][16];

    const int strip = blockIdx.x;          // row tile index
    const int tid   = threadIdx.x;
    const int wave  = tid >> 5;
    const int lane  = tid & 31;
    const int l15   = lane & 15;
    const int hsel  = (lane & 16) ? 8 : 0; // per-half K sub-offset (ISA A/B layout)
    const int mbase = (lane & 16) ? 8 : 0; // per-half row group (ISA C layout)

    // Stage the 16x512 A strip into LDS, split into hi/lo bf16.
    for (int idx = tid; idx < 16 * D_K; idx += 256) {
        int m = idx >> 9;
        int k = idx & (D_K - 1);
        float x = zs[(size_t)(strip * 16 + m) * D_K + k];
        __bf16 h = (__bf16)x;
        Ah[m * LDS_STRIDE + k] = h;
        Al[m * LDS_STRIDE + k] = (__bf16)(x - (float)h);
    }
    if (tid < 16) rowlab[tid] = labels[strip * 16 + tid];
    __syncthreads();

    float rmin[8];
#pragma unroll
    for (int r = 0; r < 8; ++r) rmin[r] = BIG_F;

    const __bf16* arow_h = &Ah[l15 * LDS_STRIDE + hsel];
    const __bf16* arow_l = &Al[l15 * LDS_STRIDE + hsel];

    for (int tb = wave * TBLK; tb < NTILE; tb += 8 * TBLK) {
        // Per-tile column index / label / B row pointers.
        int          ng[TBLK];
        int          clab[TBLK];
        const __bf16* bh[TBLK];
        const __bf16* bl[TBLK];
#pragma unroll
        for (int t = 0; t < TBLK; ++t) {
            ng[t]   = (tb + t) * 16 + l15;
            clab[t] = labels[ng[t]];
            bh[t]   = zih + (size_t)ng[t] * D_K + hsel;
            bl[t]   = zil + (size_t)ng[t] * D_K + hsel;
        }

        // Prefetch next tile group's B rows while this group computes.
        if (tb + 8 * TBLK < NTILE) {
            const size_t nb = (size_t)((tb + 8 * TBLK) * 16 + l15) * D_K;
            __builtin_prefetch(zih + nb, 0, 1);
            __builtin_prefetch(zil + nb, 0, 1);
        }

        v8f acc0 = {}, acc1 = {}, acc2 = {}, acc3 = {};
#pragma unroll 2
        for (int c = 0; c < KCHUNKS; ++c) {
            const int k0 = c * 32;
            // A fragments: loaded once per chunk, reused by all 4 tiles.
            v16bf afh = cat16(*(const v8bf*)(arow_h + k0),
                              *(const v8bf*)(arow_h + k0 + 16));
            v16bf afl = cat16(*(const v8bf*)(arow_l + k0),
                              *(const v8bf*)(arow_l + k0 + 16));

            v16bf bfh0 = cat16(*(const v8bf*)(bh[0] + k0), *(const v8bf*)(bh[0] + k0 + 16));
            v16bf bfl0 = cat16(*(const v8bf*)(bl[0] + k0), *(const v8bf*)(bl[0] + k0 + 16));
            v16bf bfh1 = cat16(*(const v8bf*)(bh[1] + k0), *(const v8bf*)(bh[1] + k0 + 16));
            v16bf bfl1 = cat16(*(const v8bf*)(bl[1] + k0), *(const v8bf*)(bl[1] + k0 + 16));
            v16bf bfh2 = cat16(*(const v8bf*)(bh[2] + k0), *(const v8bf*)(bh[2] + k0 + 16));
            v16bf bfl2 = cat16(*(const v8bf*)(bl[2] + k0), *(const v8bf*)(bl[2] + k0 + 16));
            v16bf bfh3 = cat16(*(const v8bf*)(bh[3] + k0), *(const v8bf*)(bh[3] + k0 + 16));
            v16bf bfl3 = cat16(*(const v8bf*)(bl[3] + k0), *(const v8bf*)(bl[3] + k0 + 16));

            // x*y ~= hi*hi + hi*lo + lo*hi (compensated bf16); 4 independent
            // accumulation chains keep the matrix pipe busy.
            acc0 = __builtin_amdgcn_wmma_f32_16x16x32_bf16(false, afh, false, bfh0, (short)0, acc0, false, false);
            acc1 = __builtin_amdgcn_wmma_f32_16x16x32_bf16(false, afh, false, bfh1, (short)0, acc1, false, false);
            acc2 = __builtin_amdgcn_wmma_f32_16x16x32_bf16(false, afh, false, bfh2, (short)0, acc2, false, false);
            acc3 = __builtin_amdgcn_wmma_f32_16x16x32_bf16(false, afh, false, bfh3, (short)0, acc3, false, false);
            acc0 = __builtin_amdgcn_wmma_f32_16x16x32_bf16(false, afh, false, bfl0, (short)0, acc0, false, false);
            acc1 = __builtin_amdgcn_wmma_f32_16x16x32_bf16(false, afh, false, bfl1, (short)0, acc1, false, false);
            acc2 = __builtin_amdgcn_wmma_f32_16x16x32_bf16(false, afh, false, bfl2, (short)0, acc2, false, false);
            acc3 = __builtin_amdgcn_wmma_f32_16x16x32_bf16(false, afh, false, bfl3, (short)0, acc3, false, false);
            acc0 = __builtin_amdgcn_wmma_f32_16x16x32_bf16(false, afl, false, bfh0, (short)0, acc0, false, false);
            acc1 = __builtin_amdgcn_wmma_f32_16x16x32_bf16(false, afl, false, bfh1, (short)0, acc1, false, false);
            acc2 = __builtin_amdgcn_wmma_f32_16x16x32_bf16(false, afl, false, bfh2, (short)0, acc2, false, false);
            acc3 = __builtin_amdgcn_wmma_f32_16x16x32_bf16(false, afl, false, bfh3, (short)0, acc3, false, false);
        }

        // Epilogue: dists = 1 - acc; mask same-label, row-min across 16 lanes.
#pragma unroll
        for (int t = 0; t < TBLK; ++t) {
            const v8f acc = (t == 0) ? acc0 : (t == 1) ? acc1 : (t == 2) ? acc2 : acc3;
#pragma unroll
            for (int r = 0; r < 8; ++r) {
                const int ml = mbase + r;                  // local row 0..15
                float d = 1.0f - acc[r];
                if (strip == tb + t && ml == l15) pdiag[ml] = d;  // diagonal positive
                float cand = (clab[t] != rowlab[ml]) ? d : BIG_F;
                cand = fminf(cand, __shfl_xor(cand, 1, 32));
                cand = fminf(cand, __shfl_xor(cand, 2, 32));
                cand = fminf(cand, __shfl_xor(cand, 4, 32));
                cand = fminf(cand, __shfl_xor(cand, 8, 32)); // stays within 16-lane half
                rmin[r] = fminf(rmin[r], cand);
            }
        }
    }

    if (l15 == 0) {
#pragma unroll
        for (int r = 0; r < 8; ++r) wmin[wave][mbase + r] = rmin[r];
    }
    __syncthreads();

    // First 16 threads: combine the 8 waves, compute per-anchor hinge, reduce.
    if (tid < 16) {
        float n = BIG_F;
#pragma unroll
        for (int w = 0; w < 8; ++w) n = fminf(n, wmin[w][tid]);
        const bool has_neg = n < 0.5f * BIG_F;
        float contrib = has_neg ? fmaxf(pdiag[tid] - n + MARGIN_F, 0.0f) : 0.0f;
        int   c       = has_neg ? 1 : 0;
#pragma unroll
        for (int off = 1; off < 16; off <<= 1) {
            contrib += __shfl_xor(contrib, off, 32);
            c       += __shfl_xor(c, off, 32);
        }
        if (tid == 0) {
            atomicAdd(total, contrib);
            atomicAdd(count, c);
        }
    }
}

// --------------------------- kernel 3: finalize -----------------------------
__global__ void tm_finalize(const float* total, const int* count, float* out) {
    float t = *total;
    int   c = *count;
    out[0] = (c > 0) ? (t / (float)c) : 0.0f;
}

// ---------------------------------------------------------------------------
extern "C" void kernel_launch(void* const* d_in, const int* in_sizes, int n_in,
                              void* d_out, int out_size, void* d_ws, size_t ws_size,
                              hipStream_t stream) {
    (void)in_sizes; (void)n_in; (void)out_size; (void)ws_size;

    const float* zs     = (const float*)d_in[0];
    const float* zi     = (const float*)d_in[1];
    const int*   labels = (const int*)d_in[2];
    float*       out    = (float*)d_out;

    const size_t n = (size_t)B_N * D_K;   // 2,097,152 elements per tensor
    __bf16* zsh = (__bf16*)d_ws;          // 4 MB each
    __bf16* zsl = zsh + n;
    __bf16* zih = zsl + n;
    __bf16* zil = zih + n;
    float*  total = (float*)(zil + n);
    int*    count = (int*)(total + 1);

    tm_zero_accum<<<1, 1, 0, stream>>>(total, count);
    tm_split_bf16<<<(unsigned)((2 * n + 255) / 256), 256, 0, stream>>>(
        zs, zi, zsh, zsl, zih, zil, (int)n);
    tm_gemm_mine<<<NSTRIP, 256, 0, stream>>>(zs, zih, zil, labels, total, count);
    tm_finalize<<<1, 1, 0, stream>>>(total, count, out);
}
